// TNO_21036749816384
// MI455X (gfx1250) — matrix operand order repaired and checked by reading, not compiled, not run on hardware
//
#include <hip/hip_runtime.h>
#include <stdint.h>

// ---------------------------------------------------------------------------
// TNO block for MI455X (gfx1250, wave32, WMMA, TDM/async LDS staging).
//   1) convert Wu/Wv/Wo f32->bf16, build Toeplitz coef vector (bf16)
//   2) RMSNorm(x) -> xn (bf16)
//   3) u  = silu(xn @ Wu^T + bu)            [WMMA bf16, f32 accum]
//      vT = silu(xn @ Wv^T + bv) transposed to (b,h,hd,n)
//   4) o2 = u * (T_h @ v) per (b,h)         [WMMA, causal block skip, gating]
//   5) out = o2 @ Wo^T + bo + x             [WMMA, f32 epilogue]
// Tile staging priority: TENSOR_LOAD_TO_LDS (TDM, one DMA per tile, HW pad)
//   > GLOBAL_LOAD_ASYNC_TO_LDS_B128 > plain load/ds_store.
// ---------------------------------------------------------------------------

typedef __bf16 bf16;
typedef __attribute__((ext_vector_type(16))) __bf16 v16bf;
typedef __attribute__((ext_vector_type(8)))  __bf16 v8bf;
typedef __attribute__((ext_vector_type(8)))  float  v8f;
typedef __attribute__((ext_vector_type(4)))  int    v4i;

#define LN   2048      // sequence length
#define BBATCH 4       // batch
#define EE   1024      // embed dim
#define HH   8         // heads
#define DD1  2048      // expanded dim
#define HD   256       // head dim
#define MT   (LN*BBATCH) // 8192 tokens
#define EPSF 1e-8f

#define BM 128
#define BN 128
#define BK 32
#define LSTR 40        // LDS row stride (bf16): 64B data + 16B pad, 16B aligned

#if __has_builtin(__builtin_amdgcn_tensor_load_to_lds) && \
    __has_builtin(__builtin_amdgcn_s_wait_tensorcnt)
#define HAS_TDM 1
#else
#define HAS_TDM 0
#endif

#if __has_builtin(__builtin_amdgcn_global_load_async_to_lds_b128) && \
    __has_builtin(__builtin_amdgcn_s_wait_asynccnt)
#define HAS_ASYNC 1
#else
#define HAS_ASYNC 0
#endif

#if HAS_TDM
typedef __attribute__((ext_vector_type(4))) unsigned int u32x4;
typedef __attribute__((ext_vector_type(8))) int i32x8;
typedef __attribute__((ext_vector_type(4))) int i32x4;

// One TDM descriptor: 2D tile (32 x 128 bf16), row = 64B, LDS pad 16B per row.
// D# groups per CDNA5 ISA 8.3-8.6. 6-arg builtin form (this toolchain).
__device__ __forceinline__ void tdm_load_tile(const bf16* g, bf16* lds, int ldg) {
  uint64_t ga = (uint64_t)(uintptr_t)g;
  uint32_t la = (uint32_t)(uintptr_t)lds;
  u32x4 g0;
  g0[0] = 1u;                                   // count=1, user mode, no gather
  g0[1] = la;                                   // lds_addr (bytes)
  g0[2] = (uint32_t)ga;                         // global_addr[31:0]
  g0[3] = (uint32_t)((ga >> 32) & 0x1FFFFFFu)   // global_addr[56:32]
        | (2u << 30);                           // type=2 ("image")
  i32x8 g1;
  g1[0] = (1 << 16)                             // data_size: 2 bytes
        | (1 << 20)                             // pad_enable
        | (3 << 22)                             // pad_interval: 16 DWORDs (64B)
        | (3 << 25);                            // pad_amount: 4 DWORDs (16B)
  g1[1] = (BK << 16);                           // tensor_dim0[15:0] = 32
  g1[2] = (BM << 16);                           // dim0[31:16]=0 | tensor_dim1[15:0]=128
  g1[3] = (BK << 16);                           // dim1[31:16]=0 | tile_dim0=32
  g1[4] = BM;                                   // tile_dim1=128 | tile_dim2=0
  g1[5] = (int)(uint32_t)ldg;                   // tensor_dim0_stride[31:0] (elements)
  g1[6] = 0;                                    // stride0[47:32] | dim1_stride[15:0]
  g1[7] = 0;
  i32x4 z4 = {0, 0, 0, 0};
  i32x8 z8 = {0, 0, 0, 0, 0, 0, 0, 0};
  __builtin_amdgcn_tensor_load_to_lds(g0, g1, z4, z4, z8, 0);
}
#endif

#if HAS_ASYNC
typedef __attribute__((address_space(1))) v4i as1_v4i;   // global int4
typedef __attribute__((address_space(3))) v4i as3_v4i;   // LDS int4
__device__ __forceinline__ void async_copy16(const bf16* __restrict__ g, bf16* l) {
  __builtin_amdgcn_global_load_async_to_lds_b128(
      (as1_v4i*)(uintptr_t)g, (as3_v4i*)(uint32_t)(uintptr_t)l, 0, 0);
}
#endif

// cooperative fallback loader (all 256 threads)
__device__ __forceinline__ void load_tile(bf16* lds, const bf16* __restrict__ g, int ldg) {
  int t = threadIdx.x;
#if HAS_ASYNC
#pragma unroll
  for (int it = 0; it < 2; ++it) {
    int idx = it * 256 + t;          // 512 16B chunks
    int row = idx >> 2;
    int cb  = (idx & 3) << 3;
    async_copy16(g + (size_t)row * ldg + cb, lds + row * LSTR + cb);
  }
#else
#pragma unroll
  for (int it = 0; it < 4; ++it) {
    int idx = it * 256 + t;
    int row = idx >> 3;
    int cb  = (idx & 7) << 2;
    uint2 d = *(const uint2*)(g + (size_t)row * ldg + cb);
    *(uint2*)(lds + row * LSTR + cb) = d;
  }
#endif
}

// stage one tile; TDM path: a single DMA issued by wave 0
__device__ __forceinline__ void stage_tile(bf16* lds, const bf16* __restrict__ g,
                                           int ldg, int wave) {
#if HAS_TDM
  if (wave == 0) tdm_load_tile(g, lds, ldg);
#else
  (void)wave;
  load_tile(lds, g, ldg);
#endif
}

// wait so that all tiles except the newest nT are resident (then barrier outside)
#if HAS_TDM
#define STAGE_WAIT(nT) do { if (wave == 0) __builtin_amdgcn_s_wait_tensorcnt(nT); } while (0)
#elif HAS_ASYNC
#define STAGE_WAIT(nT) __builtin_amdgcn_s_wait_asynccnt(2 * (nT))
#else
#define STAGE_WAIT(nT) do { } while (0)
#endif

union Frag { v16bf v; v8bf h[2]; };

__device__ __forceinline__ v8f wmma_bf16(v16bf a, v16bf b, v8f c) {
  return __builtin_amdgcn_wmma_f32_16x16x32_bf16(false, a, false, b, (short)0, c, false, false);
}

__device__ __forceinline__ float silu_f(float x) {
  return x * __builtin_amdgcn_rcpf(1.f + __expf(-x));
}

// A-fragment (16x32 bf16, MxK) per ISA 7.12.2.
__device__ __forceinline__ v16bf load_fragA(const bf16* lds, int rowBase, int lane) {
  Frag f;
  int r  = rowBase + (lane & 15);
  int kb = (lane >> 4) ? 8 : 0;
  const bf16* p = lds + r * LSTR + kb;
  f.h[0] = *(const v8bf*)(p);
  f.h[1] = *(const v8bf*)(p + 16);
  return f.v;
}

// B-fragment (32x16 bf16, KxN), LDS tile stored [N][K].
__device__ __forceinline__ v16bf load_fragB(const bf16* lds, int rowBase, int lane) {
  Frag f;
  int r  = rowBase + (lane & 15);
  int kb = (lane >> 4) ? 8 : 0;
  const bf16* p = lds + r * LSTR + kb;
  f.h[0] = *(const v8bf*)(p);
  f.h[1] = *(const v8bf*)(p + 16);
  return f.v;
}

// ---------------------------------------------------------------------------
__global__ __launch_bounds__(256)
void convert_kernel(const float* __restrict__ src, bf16* __restrict__ dst, int n) {
  int i = blockIdx.x * 256 + threadIdx.x;
  if (i < n) dst[i] = (bf16)src[i];
}

__global__ __launch_bounds__(256)
void coef_kernel(const float* __restrict__ zero, const float* __restrict__ pos,
                 bf16* __restrict__ coef) {
  int i = blockIdx.x * 256 + threadIdx.x;   // H*L entries
  if (i < HH * LN) {
    int h = i >> 11;
    int j = i & (LN - 1);
    float v = (j == 0) ? zero[h] : pos[h * (LN - 1) + j - 1];
    coef[i] = (bf16)v;
  }
}

__global__ __launch_bounds__(256)
void rmsnorm_kernel(const float* __restrict__ X, bf16* __restrict__ Xn) {
  int row = blockIdx.x;
  const float* xr = X + (size_t)row * EE;
  float v[4]; float s = 0.f;
#pragma unroll
  for (int j = 0; j < 4; ++j) { v[j] = xr[threadIdx.x + j * 256]; s += v[j] * v[j]; }
#pragma unroll
  for (int o = 16; o > 0; o >>= 1) s += __shfl_xor(s, o, 32);
  __shared__ float red[8];
  int lane = threadIdx.x & 31, wave = threadIdx.x >> 5;
  if (lane == 0) red[wave] = s;
  __syncthreads();
  float tot = 0.f;
#pragma unroll
  for (int w = 0; w < 8; ++w) tot += red[w];
  float scale = 1.f / (sqrtf(tot * (1.f / EE)) + EPSF);
  bf16* xo = Xn + (size_t)row * EE;
#pragma unroll
  for (int j = 0; j < 4; ++j) xo[threadIdx.x + j * 256] = (bf16)(v[j] * scale);
}

// ---------------------------------------------------------------------------
// C[M=8192, N=2048] = silu(xn @ W^T + bias). mode 0 -> U token-major,
// mode 1 -> V transposed to (b, h, hd, n).
__global__ __launch_bounds__(256)
void gemm_silu_kernel(const bf16* __restrict__ A, const bf16* __restrict__ W,
                      const float* __restrict__ bias,
                      bf16* __restrict__ U, bf16* __restrict__ Vt, int mode) {
  __shared__ __align__(16) bf16 As[2][BM * LSTR];
  __shared__ __align__(16) bf16 Bs[2][BN * LSTR];
  const int K = EE;
  int n0 = blockIdx.x * BN;
  int m0 = blockIdx.y * BM;
  int lane = threadIdx.x & 31;
  int wave = threadIdx.x >> 5;
  int mw = (wave >> 2) * 64;
  int nw = (wave & 3) * 32;

  v8f acc[4][2];
  v8f z = {0.f,0.f,0.f,0.f,0.f,0.f,0.f,0.f};
#pragma unroll
  for (int mi = 0; mi < 4; ++mi)
#pragma unroll
    for (int ni = 0; ni < 2; ++ni) acc[mi][ni] = z;

  stage_tile(As[0], A + (size_t)m0 * K, K, wave);
  stage_tile(Bs[0], W + (size_t)n0 * K, K, wave);
  int cur = 0;
  for (int k0 = 0; k0 < K; k0 += BK) {
    if (k0 + BK < K) {
      stage_tile(As[cur ^ 1], A + (size_t)m0 * K + k0 + BK, K, wave);
      stage_tile(Bs[cur ^ 1], W + (size_t)n0 * K + k0 + BK, K, wave);
      STAGE_WAIT(2);     // 2 newer tiles in flight -> current stage resident
    } else {
      STAGE_WAIT(0);
    }
    __syncthreads();
    v16bf af[4], bfr[2];
#pragma unroll
    for (int mi = 0; mi < 4; ++mi) af[mi] = load_fragA(As[cur], mw + mi * 16, lane);
#pragma unroll
    for (int ni = 0; ni < 2; ++ni) bfr[ni] = load_fragB(Bs[cur], nw + ni * 16, lane);
#pragma unroll
    for (int mi = 0; mi < 4; ++mi)
#pragma unroll
      for (int ni = 0; ni < 2; ++ni)
        acc[mi][ni] = wmma_bf16(af[mi], bfr[ni], acc[mi][ni]);
    __syncthreads();
    cur ^= 1;
  }

#pragma unroll
  for (int mi = 0; mi < 4; ++mi) {
    int row0 = m0 + mw + mi * 16 + ((lane >> 4) ? 8 : 0);
#pragma unroll
    for (int ni = 0; ni < 2; ++ni) {
      int col = n0 + nw + ni * 16 + (lane & 15);
      float bv = bias[col];
#pragma unroll
      for (int r = 0; r < 8; ++r) {
        int row = row0 + r;
        float val = silu_f(acc[mi][ni][r] + bv);
        bf16 o = (bf16)val;
        if (mode == 0) {
          U[(size_t)row * DD1 + col] = o;
        } else {
          int n  = row >> 2;         // token = n*B + b
          int bi = row & 3;
          int h  = col >> 8;
          int hd = col & 255;
          Vt[(size_t)((bi * HH + h) * HD + hd) * LN + n] = o;
        }
      }
    }
  }
}

// ---------------------------------------------------------------------------
// Per (b,h): o[n, hd] = sum_{s<=n} coef[n-s] * v[s, hd]; gated by u.
__global__ __launch_bounds__(256)
void toeplitz_kernel(const bf16* __restrict__ coef, const bf16* __restrict__ Vt,
                     const bf16* __restrict__ U, bf16* __restrict__ O2) {
  __shared__ __align__(16) bf16 Bs[2][BN * LSTR];
  __shared__ __align__(16) bf16 Cs[LN];           // per-head Toeplitz coefs
  const int K = LN;
  int n0 = blockIdx.x * BN;       // hd tile
  int m0 = blockIdx.y * BM;       // sequence tile
  int zid = blockIdx.z;           // 0..31 = b*H + h
  int bi = zid >> 3;
  int h  = zid & 7;
  const bf16* ch = coef + h * LN;
  const bf16* Vh = Vt + (size_t)((bi * HH + h) * HD) * LN;   // rows=hd, ld=LN

  int lane = threadIdx.x & 31;
  int wave = threadIdx.x >> 5;
  int mw = (wave >> 2) * 64;
  int nw = (wave & 3) * 32;

  // stage coef vector once (only [0, m0+BM) is ever indexed)
  for (int i = threadIdx.x; i < LN; i += 256) Cs[i] = ch[i];

  v8f acc[4][2];
  v8f z = {0.f,0.f,0.f,0.f,0.f,0.f,0.f,0.f};
#pragma unroll
  for (int mi = 0; mi < 4; ++mi)
#pragma unroll
    for (int ni = 0; ni < 2; ++ni) acc[mi][ni] = z;

  int kend = m0 + BM;             // causal: K blocks beyond tile diagonal are zero
  if (kend > K) kend = K;
  stage_tile(Bs[0], Vh + (size_t)n0 * LN, LN, wave);
  int cur = 0;
  for (int k0 = 0; k0 < kend; k0 += BK) {
    if (k0 + BK < kend) {
      stage_tile(Bs[cur ^ 1], Vh + (size_t)n0 * LN + k0 + BK, LN, wave);
      STAGE_WAIT(1);
    } else {
      STAGE_WAIT(0);
    }
    __syncthreads();
    v16bf af[4], bfr[2];
#pragma unroll
    for (int mi = 0; mi < 4; ++mi) {
      int t  = m0 + mw + mi * 16 + (lane & 15);
      int sb = k0 + ((lane >> 4) ? 8 : 0);
      Frag f;
#pragma unroll
      for (int i = 0; i < 8; ++i) {
        int s = sb + i;
        f.h[0][i] = (t >= s) ? Cs[t - s] : (bf16)0.f;
      }
#pragma unroll
      for (int i = 0; i < 8; ++i) {
        int s = sb + 16 + i;
        f.h[1][i] = (t >= s) ? Cs[t - s] : (bf16)0.f;
      }
      af[mi] = f.v;
    }
#pragma unroll
    for (int ni = 0; ni < 2; ++ni) bfr[ni] = load_fragB(Bs[cur], nw + ni * 16, lane);
#pragma unroll
    for (int mi = 0; mi < 4; ++mi)
#pragma unroll
      for (int ni = 0; ni < 2; ++ni)
        acc[mi][ni] = wmma_bf16(af[mi], bfr[ni], acc[mi][ni]);
    __syncthreads();
    cur ^= 1;
  }

#pragma unroll
  for (int mi = 0; mi < 4; ++mi) {
    int nrow0 = m0 + mw + mi * 16 + ((lane >> 4) ? 8 : 0);
#pragma unroll
    for (int ni = 0; ni < 2; ++ni) {
      int hd = n0 + nw + ni * 16 + (lane & 15);
      int d  = h * HD + hd;
#pragma unroll
      for (int r = 0; r < 8; ++r) {
        int n = nrow0 + r;
        size_t idx = (size_t)(n * BBATCH + bi) * DD1 + d;
        float uval = (float)U[idx];
        O2[idx] = (bf16)(uval * acc[mi][ni][r]);
      }
    }
  }
}

// ---------------------------------------------------------------------------
// out[M=8192, E=1024] = o2 @ Wo^T + bo + x  (float output)
__global__ __launch_bounds__(256)
void gemm_out_kernel(const bf16* __restrict__ A, const bf16* __restrict__ W,
                     const float* __restrict__ bias, const float* __restrict__ X,
                     float* __restrict__ Out) {
  __shared__ __align__(16) bf16 As[2][BM * LSTR];
  __shared__ __align__(16) bf16 Bs[2][BN * LSTR];
  const int K = DD1;
  int n0 = blockIdx.x * BN;
  int m0 = blockIdx.y * BM;
  int lane = threadIdx.x & 31;
  int wave = threadIdx.x >> 5;
  int mw = (wave >> 2) * 64;
  int nw = (wave & 3) * 32;

  v8f acc[4][2];
  v8f z = {0.f,0.f,0.f,0.f,0.f,0.f,0.f,0.f};
#pragma unroll
  for (int mi = 0; mi < 4; ++mi)
#pragma unroll
    for (int ni = 0; ni < 2; ++ni) acc[mi][ni] = z;

  stage_tile(As[0], A + (size_t)m0 * K, K, wave);
  stage_tile(Bs[0], W + (size_t)n0 * K, K, wave);
  int cur = 0;
  for (int k0 = 0; k0 < K; k0 += BK) {
    if (k0 + BK < K) {
      stage_tile(As[cur ^ 1], A + (size_t)m0 * K + k0 + BK, K, wave);
      stage_tile(Bs[cur ^ 1], W + (size_t)n0 * K + k0 + BK, K, wave);
      STAGE_WAIT(2);
    } else {
      STAGE_WAIT(0);
    }
    __syncthreads();
    v16bf af[4], bfr[2];
#pragma unroll
    for (int mi = 0; mi < 4; ++mi) af[mi] = load_fragA(As[cur], mw + mi * 16, lane);
#pragma unroll
    for (int ni = 0; ni < 2; ++ni) bfr[ni] = load_fragB(Bs[cur], nw + ni * 16, lane);
#pragma unroll
    for (int mi = 0; mi < 4; ++mi)
#pragma unroll
      for (int ni = 0; ni < 2; ++ni)
        acc[mi][ni] = wmma_bf16(af[mi], bfr[ni], acc[mi][ni]);
    __syncthreads();
    cur ^= 1;
  }

#pragma unroll
  for (int mi = 0; mi < 4; ++mi) {
    int row0 = m0 + mw + mi * 16 + ((lane >> 4) ? 8 : 0);
#pragma unroll
    for (int ni = 0; ni < 2; ++ni) {
      int col = n0 + nw + ni * 16 + (lane & 15);
      float bv = bias[col];
#pragma unroll
      for (int r = 0; r < 8; ++r) {
        int row = row0 + r;
        size_t idx = (size_t)row * EE + col;
        Out[idx] = acc[mi][ni][r] + bv + X[idx];
      }
    }
  }
}

// ---------------------------------------------------------------------------
extern "C" void kernel_launch(void* const* d_in, const int* in_sizes, int n_in,
                              void* d_out, int out_size, void* d_ws, size_t ws_size,
                              hipStream_t stream) {
  const float* x    = (const float*)d_in[0];
  const float* Wu   = (const float*)d_in[1];
  const float* bu   = (const float*)d_in[2];
  const float* Wv   = (const float*)d_in[3];
  const float* bv   = (const float*)d_in[4];
  const float* Wo   = (const float*)d_in[5];
  const float* bo   = (const float*)d_in[6];
  const float* zero = (const float*)d_in[7];
  const float* pos  = (const float*)d_in[8];
  float* out = (float*)d_out;

  char* ws = (char*)d_ws;
  bf16* xn   = (bf16*)(ws + 0);            // 8192*1024*2  = 16,777,216
  bf16* wu_b = (bf16*)(ws + 16777216);     // 2048*1024*2  =  4,194,304
  bf16* wv_b = (bf16*)(ws + 20971520);
  bf16* wo_b = (bf16*)(ws + 25165824);
  bf16* coef = (bf16*)(ws + 29360128);     // 8*2048*2     =     32,768
  bf16* u_b  = (bf16*)(ws + 29392896);     // 8192*2048*2  = 33,554,432
  bf16* vt_b = (bf16*)(ws + 62947328);
  bf16* o2_b = (bf16*)(ws + 96501760);     // end: 130,056,192 bytes

  const int WN = DD1 * EE;
  convert_kernel<<<(WN + 255) / 256, 256, 0, stream>>>(Wu, wu_b, WN);
  convert_kernel<<<(WN + 255) / 256, 256, 0, stream>>>(Wv, wv_b, WN);
  convert_kernel<<<(WN + 255) / 256, 256, 0, stream>>>(Wo, wo_b, WN);
  coef_kernel<<<(HH * LN + 255) / 256, 256, 0, stream>>>(zero, pos, coef);

  rmsnorm_kernel<<<MT, 256, 0, stream>>>(x, xn);

  dim3 g1(DD1 / BN, MT / BM);              // 16 x 64
  gemm_silu_kernel<<<g1, 256, 0, stream>>>(xn, wu_b, bu, u_b, vt_b, 0);
  gemm_silu_kernel<<<g1, 256, 0, stream>>>(xn, wv_b, bv, u_b, vt_b, 1);

  dim3 g2(HD / BN, LN / BM, BBATCH * HH);  // 2 x 16 x 32
  toeplitz_kernel<<<g2, 256, 0, stream>>>(coef, vt_b, u_b, o2_b);

  dim3 g3(EE / BN, MT / BM);               // 8 x 64
  gemm_out_kernel<<<g3, 256, 0, stream>>>(o2_b, wo_b, bo, x, out);
}
